// USMSharp_66468913873012
// MI455X (gfx1250) — matrix operand order repaired and checked by reading, not compiled, not run on hardware
//
#include <hip/hip_runtime.h>

typedef __attribute__((ext_vector_type(16))) _Float16 v16h;
typedef __attribute__((ext_vector_type(8)))  float    v8f;

#define HH   1080
#define WW   1920
#define NPL  6        // B*C planes
#define RAD  25       // 51-tap kernel, 25 each side
#define TAPS 51
#define TILE 64
#define INR  128      // staged input rows (64 out + 2*25 halo, padded to 8x16)
#define INC  144      // staged input cols (covers K-padding reads up to 48+95)
#define INP  146      // input LDS pitch (halfs) - odd dword count
#define HR   144      // h-scratch rows (covers vconv K-padding reads)
#define HP   74       // h-scratch pitch (halfs) - odd dword count

__device__ __forceinline__ int refl(int i, int n) {
    if (i < 0)  i = -i;           // jnp 'reflect' (mirror w/o edge repeat)
    if (i >= n) i = 2 * n - 2 - i;
    return i;
}

// PASS 0: blur(img) -> mask(u8), delta = (sharp - img) as f16
// PASS 1: blur(mask) -> out = img + soft_mask * delta
template <int PASS>
__global__ __launch_bounds__(256) void usm_pass(
    const float* __restrict__ img, const float* __restrict__ k2d,
    unsigned char* __restrict__ mask, _Float16* __restrict__ delta,
    float* __restrict__ out)
{
    __shared__ _Float16 s_in[INR * INP];   // 37376 B
    __shared__ _Float16 s_h[HR * HP];      // 21312 B

    const int plane = blockIdx.z;
    const int x0 = blockIdx.x * TILE;
    const int y0 = blockIdx.y * TILE;
    const size_t pbase = (size_t)plane * HH * WW;
    const float* src = img + pbase;

    const int lane = threadIdx.x & 31;
    const int hf = lane >> 4;     // lane half (K-group select)
    const int ln = lane & 15;     // row (A) / col (B,D) within fragment
    const int wv = threadIdx.x >> 5;

    // ---- 1D Gaussian weights as WMMA band fragments (registers) ----
    // k1[i] = k2d[i][25] * rsqrt(k2d[25][25])  since k2d = outer(k1,k1)
    const float rs = rsqrtf(k2d[25 * TAPS + 25]);
    v16h bw[3];  // B-operand band for horizontal pass: B[k][j] = w1d[k - j]
    v16h aw[3];  // A-operand band for vertical pass:   A[i][k] = w1d[k - i]
#pragma unroll
    for (int q = 0; q < 3; ++q) {
        v16h b, a;
#pragma unroll
        for (int e = 0; e < 16; ++e) {
            int kb = q * 32 + hf * 16 + e;                         // B layout K
            int ib = kb - ln;
            b[e] = (ib >= 0 && ib < TAPS) ? (_Float16)(k2d[ib * TAPS + 25] * rs)
                                          : (_Float16)0.0f;
            int ka = q * 32 + hf * 16 + e;                         // A layout K
            int ia = ka - ln;
            a[e] = (ia >= 0 && ia < TAPS) ? (_Float16)(k2d[ia * TAPS + 25] * rs)
                                          : (_Float16)0.0f;
        }
        bw[q] = b; aw[q] = a;
    }

    // ---- stage input tile + halo into LDS as f16 (reflect padding) ----
    for (int t = threadIdx.x; t < INR * INC; t += 256) {
        int r = t / INC, c = t - (t / INC) * INC;
        int gy = refl(y0 + r - RAD, HH);
        int gx = refl(x0 + c - RAD, WW);
        float v;
        if (PASS == 0) v = src[(size_t)gy * WW + gx];
        else           v = (float)mask[pbase + (size_t)gy * WW + gx];
        s_in[r * INP + c] = (_Float16)v;
    }
    // zero the K-padding tail rows of the h-scratch (multiplied by zero weights,
    // but values must be finite)
    for (int t = threadIdx.x; t < (HR - INR) * HP; t += 256)
        s_h[INR * HP + t] = (_Float16)0.0f;
    __syncthreads();

    // ---- horizontal pass: h[r][j0+j] = sum_m in[r][j0+m] * w1d[m-j] ----
    // 8 row-tiles x 4 col-tiles, 3 K-chunks each, spread over 8 waves
    for (int t = wv; t < 32; t += 8) {
        int rt = t >> 2, jt = t & 3;
        v8f acc = {};
#pragma unroll
        for (int q = 0; q < 3; ++q) {
            v16h aF;
#pragma unroll
            for (int e = 0; e < 16; ++e) {
                int k = q * 32 + hf * 16 + e;
                aF[e] = s_in[(rt * 16 + ln) * INP + jt * 16 + k];
            }
            acc = __builtin_amdgcn_wmma_f32_16x16x32_f16(
                false, aF, false, bw[q], (short)0, acc, false, false);
        }
#pragma unroll
        for (int r8 = 0; r8 < 8; ++r8)
            s_h[(rt * 16 + r8 + 8 * hf) * HP + jt * 16 + ln] = (_Float16)acc[r8];
    }
    __syncthreads();

    // ---- vertical pass + epilogue: out[i0+i][c] = sum_m w1d[m-i] * h[i0+m][c] ----
    for (int t = wv; t < 16; t += 8) {
        int it = t >> 2, ct = t & 3;
        v8f acc = {};
#pragma unroll
        for (int q = 0; q < 3; ++q) {
            v16h bF;
#pragma unroll
            for (int e = 0; e < 16; ++e) {
                int k = q * 32 + hf * 16 + e;
                bF[e] = s_h[(it * 16 + k) * HP + ct * 16 + ln];
            }
            acc = __builtin_amdgcn_wmma_f32_16x16x32_f16(
                false, aw[q], false, bF, (short)0, acc, false, false);
        }
        const int gx = x0 + ct * 16 + ln;
#pragma unroll
        for (int r8 = 0; r8 < 8; ++r8) {
            int gy = y0 + it * 16 + r8 + 8 * hf;
            if (gy < HH) {
                size_t gidx = pbase + (size_t)gy * WW + gx;
                float conv = acc[r8];
                if (PASS == 0) {
                    float iv  = img[gidx];
                    float res = iv - conv;                       // residual
                    float mk  = (fabsf(res) * 255.0f > 10.0f) ? 1.0f : 0.0f;
                    float sh  = fminf(fmaxf(iv + 0.5f * res, 0.0f), 1.0f);
                    mask[gidx]  = (unsigned char)mk;
                    delta[gidx] = (_Float16)(sh - iv);
                } else {
                    float iv = img[gidx];
                    out[gidx] = iv + conv * (float)delta[gidx];  // img + sm*(sharp-img)
                }
            }
        }
    }
}

extern "C" void kernel_launch(void* const* d_in, const int* in_sizes, int n_in,
                              void* d_out, int out_size, void* d_ws, size_t ws_size,
                              hipStream_t stream)
{
    const float* img = (const float*)d_in[0];
    const float* k2d = (const float*)d_in[1];
    float* out = (float*)d_out;

    const size_t NP = (size_t)NPL * HH * WW;           // 12,441,600 px
    unsigned char* mask = (unsigned char*)d_ws;        // NP bytes
    _Float16* delta = (_Float16*)((char*)d_ws + NP);   // NP * 2 bytes (16B-aligned)

    dim3 grid(WW / TILE, (HH + TILE - 1) / TILE, NPL); // 30 x 17 x 6
    dim3 block(256);                                   // 8 waves (wave32)
    usm_pass<0><<<grid, block, 0, stream>>>(img, k2d, mask, delta, out);
    usm_pass<1><<<grid, block, 0, stream>>>(img, k2d, mask, delta, out);
}
